// BLinear_35510789603654
// MI455X (gfx1250) — compile-verified, matrix operand order checked by reading, not run on hardware
//
#include <hip/hip_runtime.h>

// ---------------------------------------------------------------------------
// BLinear: y = x @ sign(W)^T / sqrt(K)
//   x: [8192, 4096] f32   W: [4096, 4096] f32   y: [8192, 4096] f32
// Two-pass: (1) convert x->bf16, W->(+-2^-6) bf16 into workspace (scale
// 1/sqrt(4096)=2^-6 folded into binarized weight, exact in bf16).
// (2) bf16 WMMA GEMM; global->LDS staging via TDM tensor_load_to_lds (with
// hardware row padding) when available, else vector loads + ds stores.
// Fallback fused kernel used if workspace is too small.
// ---------------------------------------------------------------------------

typedef __attribute__((ext_vector_type(16))) __bf16       v16bf;
typedef __attribute__((ext_vector_type(8)))  float        v8f;
typedef __attribute__((ext_vector_type(2)))  float        v2f;
typedef __attribute__((ext_vector_type(2)))  __bf16       v2bf;
typedef __attribute__((ext_vector_type(4)))  unsigned int u32x4;
typedef __attribute__((ext_vector_type(8)))  int          i32x8;
typedef __attribute__((ext_vector_type(4)))  int          i32x4;

#define M_DIM 8192
#define N_DIM 4096
#define K_DIM 4096

#define BM 128
#define BN 128
#define BK 32
#define LDS_STRIDE 40              // 32 bf16 payload + 8 bf16 pad -> 80B rows
#define LDS_BUF_BYTES (BM * LDS_STRIDE * 2)

#if __has_builtin(__builtin_amdgcn_tensor_load_to_lds) && \
    __has_builtin(__builtin_amdgcn_s_wait_tensorcnt)
#define HAVE_TDM 1
#else
#define HAVE_TDM 0
#endif

union FragU {
  uint4 q[2];
  v16bf v;
};

// f32 pair -> packed bf16 pair (hardware cvt, RNE)
__device__ __forceinline__ unsigned cvt2_bf16(float a, float b) {
  v2f v = {a, b};
  v2bf r = __builtin_convertvector(v, v2bf);
  return __builtin_bit_cast(unsigned, r);
}

// sign(w)/64 pair in bf16: +2^-6 = 0x3C80, -2^-6 = 0xBC80 (w==0 -> -1)
__device__ __forceinline__ unsigned sign2_bf16(float a, float b) {
  unsigned lo = (a > 0.0f) ? 0x3C80u : 0xBC80u;
  unsigned hi = (b > 0.0f) ? 0x3C80u : 0xBC80u;
  return lo | (hi << 16);
}

// ---------------------------------------------------------------------------
// Pass 1: streaming conversion kernels
// ---------------------------------------------------------------------------
__global__ void cvt_x_bf16(const float4* __restrict__ X, uint2* __restrict__ O, int n4) {
  const int stride = gridDim.x * blockDim.x;
  for (int i = blockIdx.x * blockDim.x + threadIdx.x; i < n4; i += stride) {
    float4 v = X[i];
    uint2 o;
    o.x = cvt2_bf16(v.x, v.y);
    o.y = cvt2_bf16(v.z, v.w);
    O[i] = o;
  }
}

__global__ void cvt_w_sign(const float4* __restrict__ W, uint2* __restrict__ O, int n4) {
  const int stride = gridDim.x * blockDim.x;
  for (int i = blockIdx.x * blockDim.x + threadIdx.x; i < n4; i += stride) {
    float4 v = W[i];
    uint2 o;
    o.x = sign2_bf16(v.x, v.y);
    o.y = sign2_bf16(v.z, v.w);
    O[i] = o;
  }
}

// ---------------------------------------------------------------------------
// TDM descriptor: load a 128-row x 32-col bf16 tile (row stride K elements)
// into LDS with 16B padding after every 64B row -> 80B LDS rows.
// Toolchain here exposes the 6-arg builtin:
//   (u32x4 g0, i32x8 g1, i32x4 g2, i32x4 g3, i32x8 extra, i32 cpol)
// ---------------------------------------------------------------------------
#if HAVE_TDM
__device__ __forceinline__ void tdm_load_tile(unsigned long long gaddr,
                                              unsigned lds_addr,
                                              unsigned tensor_rows) {
  u32x4 g0;
  g0[0] = 1u;                                                    // count=1, user desc
  g0[1] = lds_addr;                                              // LDS byte address
  g0[2] = (unsigned)(gaddr & 0xFFFFFFFFull);                     // global_addr[31:0]
  g0[3] = (unsigned)((gaddr >> 32) & 0x1FFFFFFull) | (2u << 30); // [56:32] | type=2

  i32x8 g1;
  g1[0] = (int)((1u << 16) |                 // data_size = 2 bytes
                (1u << 20) |                 // pad_enable
                (3u << 22) |                 // pad_interval: 16 DWORDs (64B row)
                (3u << 25));                 // pad_amount:   4 DWORDs (16B)
  g1[1] = (int)(((unsigned)K_DIM & 0xFFFFu) << 16);              // tensor_dim0 lo
  g1[2] = (int)(((tensor_rows & 0xFFFFu) << 16) |
                (((unsigned)K_DIM >> 16) & 0xFFFFu));            // dim1 lo | dim0 hi
  g1[3] = (int)(((unsigned)BK << 16) |
                ((tensor_rows >> 16) & 0xFFFFu));                // tile_dim0=32 | dim1 hi
  g1[4] = (int)(unsigned)BM;                                     // tile_dim1=128, tile_dim2=0
  g1[5] = (int)(unsigned)K_DIM;                                  // tensor_dim0_stride lo
  g1[6] = 0;
  g1[7] = 0;

  i32x4 z4 = {0, 0, 0, 0};
  i32x8 z8 = {0, 0, 0, 0, 0, 0, 0, 0};
  __builtin_amdgcn_tensor_load_to_lds(g0, g1, z4, z4, z8, 0);
}
#endif

// ---------------------------------------------------------------------------
// Pass 2: bf16 WMMA GEMM from pre-converted operands
// ---------------------------------------------------------------------------
__global__ __launch_bounds__(256)
void blinear_gemm_bf16(const unsigned short* __restrict__ Xb,
                       const unsigned short* __restrict__ Wb,
                       float* __restrict__ Out) {
  __shared__ __align__(16) unsigned short As[2][BM * LDS_STRIDE];
  __shared__ __align__(16) unsigned short Bs[2][BN * LDS_STRIDE];

  const int tid   = threadIdx.x;
  const int wave  = tid >> 5;
  const int lane  = tid & 31;
  const int waveM = wave >> 2;   // 0..1
  const int waveN = wave & 3;    // 0..3
  const int half  = lane >> 4;
  const int r     = lane & 15;

  const int blockN0 = blockIdx.x * BN;
  const int blockM0 = blockIdx.y * BM;

  v8f acc[4][2];
#pragma unroll
  for (int m = 0; m < 4; ++m)
#pragma unroll
    for (int n = 0; n < 2; ++n)
#pragma unroll
      for (int e = 0; e < 8; ++e) acc[m][n][e] = 0.0f;

#if HAVE_TDM
  const unsigned ldsA0 = (unsigned)(size_t)(&As[0][0]);
  const unsigned ldsB0 = (unsigned)(size_t)(&Bs[0][0]);
  const unsigned long long gA =
      (unsigned long long)(size_t)(Xb + (size_t)blockM0 * K_DIM);
  const unsigned long long gB =
      (unsigned long long)(size_t)(Wb + (size_t)blockN0 * K_DIM);
  if (wave == 0) {
    tdm_load_tile(gA, ldsA0, (unsigned)M_DIM);
    tdm_load_tile(gB, ldsB0, (unsigned)N_DIM);
    __builtin_amdgcn_s_wait_tensorcnt(0);
  }
#else
  {
    // manual prologue: stage K-slab 0 into buffer 0
#pragma unroll
    for (int i = 0; i < 2; ++i) {
      const int id  = tid + 256 * i;    // 512 chunks of 16B per operand tile
      const int row = id >> 2;
      const int c   = id & 3;
      uint4 va = *(const uint4*)(Xb + (size_t)(blockM0 + row) * K_DIM + c * 8);
      uint4 vb = *(const uint4*)(Wb + (size_t)(blockN0 + row) * K_DIM + c * 8);
      *(uint4*)(&As[0][row * LDS_STRIDE + c * 8]) = va;
      *(uint4*)(&Bs[0][row * LDS_STRIDE + c * 8]) = vb;
    }
  }
#endif

  const int NK = K_DIM / BK;   // 128 steps
#pragma unroll 1
  for (int kk = 0; kk < NK; ++kk) {
    const int buf = kk & 1;
    __syncthreads();  // slab(buf) visible; previous consumers of buf^1 done

#if HAVE_TDM
    if (wave == 0 && kk + 1 < NK) {
      const unsigned off = (unsigned)(kk + 1) * (BK * 2);   // bytes along K
      const unsigned lb  = (unsigned)(buf ^ 1) * LDS_BUF_BYTES;
      tdm_load_tile(gA + off, ldsA0 + lb, (unsigned)M_DIM);
      tdm_load_tile(gB + off, ldsB0 + lb, (unsigned)N_DIM);
    }
#else
    if (kk + 1 < NK) {
      const int b  = buf ^ 1;
      const int k0 = (kk + 1) * BK;
#pragma unroll
      for (int i = 0; i < 2; ++i) {
        const int id  = tid + 256 * i;
        const int row = id >> 2;
        const int c   = id & 3;
        uint4 va = *(const uint4*)(Xb + (size_t)(blockM0 + row) * K_DIM + k0 + c * 8);
        uint4 vb = *(const uint4*)(Wb + (size_t)(blockN0 + row) * K_DIM + k0 + c * 8);
        *(uint4*)(&As[b][row * LDS_STRIDE + c * 8]) = va;
        *(uint4*)(&Bs[b][row * LDS_STRIDE + c * 8]) = vb;
      }
    }
#endif

    // B fragments: lanes index N; K packing matches ISA 16-bit operand layout
    FragU bfr[2];
#pragma unroll
    for (int n = 0; n < 2; ++n) {
      const int nr = waveN * 32 + n * 16 + r;
      bfr[n].q[0] = *(const uint4*)(&Bs[buf][nr * LDS_STRIDE + half * 8]);
      bfr[n].q[1] = *(const uint4*)(&Bs[buf][nr * LDS_STRIDE + 16 + half * 8]);
    }

#pragma unroll
    for (int m = 0; m < 4; ++m) {
      const int mr = waveM * 64 + m * 16 + r;
      FragU afr;
      afr.q[0] = *(const uint4*)(&As[buf][mr * LDS_STRIDE + half * 8]);
      afr.q[1] = *(const uint4*)(&As[buf][mr * LDS_STRIDE + 16 + half * 8]);
#pragma unroll
      for (int n = 0; n < 2; ++n) {
        acc[m][n] = __builtin_amdgcn_wmma_f32_16x16x32_bf16(
            false, afr.v, false, bfr[n].v, (short)0, acc[m][n], false, false);
      }
    }

#if HAVE_TDM
    if (wave == 0) __builtin_amdgcn_s_wait_tensorcnt(0);  // next slab landed
#endif
  }

  // C 16x16 f32 layout: VGPR v -> M = v + 8*half, N = r
#pragma unroll
  for (int m = 0; m < 4; ++m) {
    const int mrow0 = blockM0 + waveM * 64 + m * 16 + half * 8;
#pragma unroll
    for (int n = 0; n < 2; ++n) {
      const int ncol = blockN0 + waveN * 32 + n * 16 + r;
      float* o = Out + (size_t)mrow0 * N_DIM + ncol;
#pragma unroll
      for (int v = 0; v < 8; ++v) o[(size_t)v * N_DIM] = acc[m][n][v];
    }
  }
}

// ---------------------------------------------------------------------------
// Fallback: fused convert-in-loop kernel (used only if workspace too small)
// ---------------------------------------------------------------------------
__global__ __launch_bounds__(256)
void blinear_fused(const float* __restrict__ X,
                   const float* __restrict__ W,
                   float* __restrict__ Out) {
  __shared__ __align__(16) unsigned short As[2][BM * LDS_STRIDE];
  __shared__ __align__(16) unsigned short Bs[2][BN * LDS_STRIDE];

  const int tid   = threadIdx.x;
  const int wave  = tid >> 5;
  const int lane  = tid & 31;
  const int waveM = wave >> 2;
  const int waveN = wave & 3;
  const int half  = lane >> 4;
  const int r     = lane & 15;

  const int blockN0 = blockIdx.x * BN;
  const int blockM0 = blockIdx.y * BM;

  v8f acc[4][2];
#pragma unroll
  for (int m = 0; m < 4; ++m)
#pragma unroll
    for (int n = 0; n < 2; ++n)
#pragma unroll
      for (int e = 0; e < 8; ++e) acc[m][n][e] = 0.0f;

  float4 ga[4], gb[4];
#pragma unroll
  for (int i = 0; i < 4; ++i) {
    const int linear = tid + 256 * i;
    const int row = linear >> 3;
    const int c   = linear & 7;
    ga[i] = *(const float4*)(X + (size_t)(blockM0 + row) * K_DIM + c * 4);
    gb[i] = *(const float4*)(W + (size_t)(blockN0 + row) * K_DIM + c * 4);
  }

  const int NK = K_DIM / BK;
#pragma unroll 1
  for (int kk = 0; kk < NK; ++kk) {
    const int buf = kk & 1;
#pragma unroll
    for (int i = 0; i < 4; ++i) {
      const int linear = tid + 256 * i;
      const int row = linear >> 3;
      const int c   = linear & 7;
      uint2 av; av.x = cvt2_bf16(ga[i].x, ga[i].y); av.y = cvt2_bf16(ga[i].z, ga[i].w);
      uint2 bv; bv.x = sign2_bf16(gb[i].x, gb[i].y); bv.y = sign2_bf16(gb[i].z, gb[i].w);
      *(uint2*)(&As[buf][row * LDS_STRIDE + c * 4]) = av;
      *(uint2*)(&Bs[buf][row * LDS_STRIDE + c * 4]) = bv;
    }
    if (kk + 1 < NK) {
      const int k0 = (kk + 1) * BK;
#pragma unroll
      for (int i = 0; i < 4; ++i) {
        const int linear = tid + 256 * i;
        const int row = linear >> 3;
        const int c   = linear & 7;
        ga[i] = *(const float4*)(X + (size_t)(blockM0 + row) * K_DIM + k0 + c * 4);
        gb[i] = *(const float4*)(W + (size_t)(blockN0 + row) * K_DIM + k0 + c * 4);
      }
    }
    __syncthreads();

    FragU bfr[2];
#pragma unroll
    for (int n = 0; n < 2; ++n) {
      const int nr = waveN * 32 + n * 16 + r;
      bfr[n].q[0] = *(const uint4*)(&Bs[buf][nr * LDS_STRIDE + half * 8]);
      bfr[n].q[1] = *(const uint4*)(&Bs[buf][nr * LDS_STRIDE + 16 + half * 8]);
    }
#pragma unroll
    for (int m = 0; m < 4; ++m) {
      const int mr = waveM * 64 + m * 16 + r;
      FragU afr;
      afr.q[0] = *(const uint4*)(&As[buf][mr * LDS_STRIDE + half * 8]);
      afr.q[1] = *(const uint4*)(&As[buf][mr * LDS_STRIDE + 16 + half * 8]);
#pragma unroll
      for (int n = 0; n < 2; ++n) {
        acc[m][n] = __builtin_amdgcn_wmma_f32_16x16x32_bf16(
            false, afr.v, false, bfr[n].v, (short)0, acc[m][n], false, false);
      }
    }
  }

#pragma unroll
  for (int m = 0; m < 4; ++m) {
    const int mrow0 = blockM0 + waveM * 64 + m * 16 + half * 8;
#pragma unroll
    for (int n = 0; n < 2; ++n) {
      const int ncol = blockN0 + waveN * 32 + n * 16 + r;
      float* o = Out + (size_t)mrow0 * N_DIM + ncol;
#pragma unroll
      for (int v = 0; v < 8; ++v) o[(size_t)v * N_DIM] = acc[m][n][v];
    }
  }
}

// ---------------------------------------------------------------------------
extern "C" void kernel_launch(void* const* d_in, const int* in_sizes, int n_in,
                              void* d_out, int out_size, void* d_ws, size_t ws_size,
                              hipStream_t stream) {
  (void)in_sizes; (void)n_in; (void)out_size;
  const float* X = (const float*)d_in[0];   // [8192, 4096] f32
  const float* W = (const float*)d_in[1];   // [4096, 4096] f32
  float* Out = (float*)d_out;               // [8192, 4096] f32

  const size_t xb_elems = (size_t)M_DIM * K_DIM;
  const size_t wb_elems = (size_t)N_DIM * K_DIM;
  const size_t need = (xb_elems + wb_elems) * sizeof(unsigned short);

  dim3 grid(N_DIM / BN, M_DIM / BM);        // 32 x 64 blocks

  if (d_ws != nullptr && ws_size >= need) {
    unsigned short* Xb = (unsigned short*)d_ws;
    unsigned short* Wb = Xb + xb_elems;
    cvt_x_bf16<<<4096, 256, 0, stream>>>((const float4*)X, (uint2*)Xb,
                                         (int)(xb_elems / 4));
    cvt_w_sign<<<4096, 256, 0, stream>>>((const float4*)W, (uint2*)Wb,
                                         (int)(wb_elems / 4));
    blinear_gemm_bf16<<<grid, 256, 0, stream>>>(Xb, Wb, Out);
  } else {
    blinear_fused<<<grid, 256, 0, stream>>>(X, W, Out);
  }
}
